// Model_65841848648376
// MI455X (gfx1250) — matrix operand order repaired and checked by reading, not compiled, not run on hardware
//
#include <hip/hip_runtime.h>
#include <hip/hip_bf16.h>
#include <stdint.h>

// ---------------- CDNA5 WMMA / TDM types ----------------
typedef __attribute__((ext_vector_type(16))) __bf16 v16bf;
typedef __attribute__((ext_vector_type(8)))  float  v8f;
typedef __attribute__((ext_vector_type(4)))  uint32_t u32x4;
typedef __attribute__((ext_vector_type(8)))  int      i32x8;
typedef __attribute__((ext_vector_type(4)))  int      i32x4;

#define WAVES          4
#define BLK            (WAVES * 32)
#define ROWS_PER_WAVE  16
#define ROWS_PER_BLK   (WAVES * ROWS_PER_WAVE)
#define DMODEL         256
#define PVOC           1009
#define NTOK           (65536 * 2)

// round-to-nearest-even f32 -> bf16
__device__ __forceinline__ uint16_t f2bf(float f) {
  union { float f; uint32_t u; } c; c.f = f;
  uint32_t r = c.u + 0x7FFFu + ((c.u >> 16) & 1u);
  return (uint16_t)(r >> 16);
}
__device__ __forceinline__ float bf2f(uint16_t b) {
  union { uint32_t u; float f; } c; c.u = ((uint32_t)b) << 16;
  return c.f;
}

// A-fragment (16x32 bf16, M x K) from LDS row-major [16][pitch]:
// lane L: row m = L&15, half h2 = L>>4; VGPR 0..3 hold K = h2*8 + {0..7},
// VGPR 4..7 hold K = 16 + h2*8 + {0..7}  -> two contiguous 16B runs.
__device__ __forceinline__ v16bf load_frag_a_lds(const uint16_t* base, int pitch,
                                                 int kc, int lane) {
  int m = lane & 15, h2 = lane >> 4;
  const uint16_t* p = base + m * pitch + kc * 32 + h2 * 8;
  union { v16bf v; uint4 q[2]; } f;
  f.q[0] = *(const uint4*)p;
  f.q[1] = *(const uint4*)(p + 16);
  return f.v;
}
// B-fragment (32x16 bf16, K x N) from pre-swizzled pack: 16 contiguous bf16 / lane
__device__ __forceinline__ v16bf load_frag_b(const uint16_t* pack, int tileIdx, int lane) {
  const uint16_t* p = pack + ((size_t)tileIdx * 32 + lane) * 16;
  union { v16bf v; uint4 q[2]; } f;
  f.q[0] = *(const uint4*)p;
  f.q[1] = *(const uint4*)(p + 8);
  return f.v;
}
__device__ __forceinline__ v8f wmma_bf16(v16bf a, v16bf b, v8f c) {
  return __builtin_amdgcn_wmma_f32_16x16x32_bf16(false, a, false, b, (short)0, c,
                                                 false, false);
}
// 16x16 output tile, K=256 reduction (8 chunks of 32)
__device__ __forceinline__ v8f mm16_k8(const uint16_t* aBase, int pitch,
                                       const uint16_t* bPack, int nt, int lane) {
  v8f acc = {0.f,0.f,0.f,0.f,0.f,0.f,0.f,0.f};
  #pragma unroll
  for (int kc = 0; kc < 8; ++kc) {
    v16bf a = load_frag_a_lds(aBase, pitch, kc, lane);
    v16bf b = load_frag_b(bPack, nt * 8 + kc, lane);
    acc = wmma_bf16(a, b, acc);
  }
  return acc;
}
// C/D layout: lane L: col n = L&15, rows m = (L>>4)*8 + i  (i = vgpr index)
__device__ __forceinline__ void store_tile_lds(uint16_t* base, int pitch, int nt, int lane,
                                               v8f acc, float bias, bool relu) {
  int n = lane & 15, r0 = (lane >> 4) * 8;
  #pragma unroll
  for (int i = 0; i < 8; ++i) {
    float v = acc[i] + bias;
    if (relu) v = fmaxf(v, 0.f);
    base[(r0 + i) * pitch + nt * 16 + n] = f2bf(v);
  }
}

// ---------------- weight pre-pack (f32 -> bf16, B-fragment swizzle) ----------------
// packed element t: tileIdx = t>>9 (= nt*KC + kc), lane = (t>>4)&31, jw = t&15
// n = nt*16 + (lane&15);  k = kc*32 + (jw&8)*2 + (lane>>4)*8 + (jw&7)
// mode 0: W_{Q,K,V} [H,D,DH] as 256x256  (k=d, n=h*64+e)
// mode 1: W_O [H,DH,D]       as 256x256  (k=h*64+e, n=d)
// mode 2: row-major [N,K] (fc1_w, fc2_w, un_w), transposed on the fly
__global__ void pack_w(const float* __restrict__ src, uint16_t* __restrict__ dst,
                       int Ktot, int Nsrc, int mode, int total) {
  int t = blockIdx.x * blockDim.x + threadIdx.x;
  if (t >= total) return;
  int within = t & 511;
  int lane = within >> 4, jw = within & 15;
  int t9 = t >> 9;
  int KC = Ktot >> 5;
  int kc = t9 % KC, nt = t9 / KC;
  int n = nt * 16 + (lane & 15);
  int k = kc * 32 + (jw & 8) * 2 + (lane >> 4) * 8 + (jw & 7);
  float v = 0.f;
  if (n < Nsrc) {
    size_t si;
    if (mode == 0)      si = (size_t)(n >> 6) * (64 * 256) + (size_t)k * 64 + (n & 63);
    else if (mode == 1) si = (size_t)(k >> 6) * (64 * 256) + (size_t)(k & 63) * 256 + n;
    else                si = (size_t)n * Ktot + k;
    v = src[si];
  }
  dst[t] = f2bf(v);
}

// ---------------- fused: embed -> QKV -> attn(seq=2) -> O -> fc1+relu -> fc2 ----------------
#define PW (4 * 4096 + 512)   // per-wave LDS elems: 4 activation bufs (16x256) + hid chunk (16x32)

__global__ __launch_bounds__(BLK) void fused_fwd(
    const int* __restrict__ x, const float* __restrict__ tok_emb,
    const float* __restrict__ pos_emb,
    const float* __restrict__ bQ, const float* __restrict__ bK,
    const float* __restrict__ bV, const float* __restrict__ bO,
    const float* __restrict__ fc1_b, const float* __restrict__ fc2_b,
    const uint16_t* __restrict__ pq, const uint16_t* __restrict__ pkw,
    const uint16_t* __restrict__ pvw, const uint16_t* __restrict__ po,
    const uint16_t* __restrict__ pf1, const uint16_t* __restrict__ pf2,
    uint16_t* __restrict__ resid) {
  __shared__ uint16_t smem[WAVES * PW];
  int lane = threadIdx.x & 31, wave = threadIdx.x >> 5;
  uint16_t* B0 = smem + wave * PW;      // h, later z
  uint16_t* B1 = B0 + 4096;             // q, later attn_out
  uint16_t* B2 = B1 + 4096;             // k
  uint16_t* B3 = B2 + 4096;             // v
  uint16_t* HC = B3 + 4096;             // 16x32 hid chunk
  int rowBase = blockIdx.x * ROWS_PER_BLK + wave * ROWS_PER_WAVE;
  int n = lane & 15;

  // ---- embed: h = tok_emb[x] + pos_emb, bf16 into LDS ----
  for (int e = lane; e < ROWS_PER_WAVE * 128; e += 32) {
    int row = e >> 7, cp = (e & 127) << 1;
    int gt = rowBase + row;
    int tok = x[gt];
    const float* te = tok_emb + (size_t)tok * DMODEL + cp;
    const float* pe = pos_emb + (size_t)(gt & 1) * DMODEL + cp;
    uint32_t pk2 = (uint32_t)f2bf(te[0] + pe[0]) |
                   ((uint32_t)f2bf(te[1] + pe[1]) << 16);
    ((uint32_t*)B0)[e] = pk2;
  }
  __syncthreads();

  // ---- QKV projections (heads folded into 256x256) ----
  #pragma unroll 1
  for (int nt = 0; nt < 16; ++nt) {
    v8f aq = mm16_k8(B0, DMODEL, pq, nt, lane);
    store_tile_lds(B1, DMODEL, nt, lane, aq, bQ[nt * 16 + n], false);
    v8f ak = mm16_k8(B0, DMODEL, pkw, nt, lane);
    store_tile_lds(B2, DMODEL, nt, lane, ak, bK[nt * 16 + n], false);
    v8f av = mm16_k8(B0, DMODEL, pvw, nt, lane);
    store_tile_lds(B3, DMODEL, nt, lane, av, bV[nt * 16 + n], false);
  }
  __syncthreads();

  // ---- attention: 32 lanes = 8 sequences x 4 heads; seq len 2 ----
  {
    int s = lane >> 2, hh = lane & 3;
    int t0 = s * 2, t1 = t0 + 1, cb = hh * 64;
    float d00 = 0, d01 = 0, d10 = 0, d11 = 0;
    for (int e = 0; e < 64; ++e) {
      float q0 = bf2f(B1[t0 * DMODEL + cb + e]), q1 = bf2f(B1[t1 * DMODEL + cb + e]);
      float k0 = bf2f(B2[t0 * DMODEL + cb + e]), k1 = bf2f(B2[t1 * DMODEL + cb + e]);
      d00 += q0 * k0; d01 += q0 * k1; d10 += q1 * k0; d11 += q1 * k1;
    }
    const float sc = 0.125f;   // 1/sqrt(64)
    d00 *= sc; d01 *= sc; d10 *= sc; d11 *= sc;
    float m0 = fmaxf(d00, d01), m1 = fmaxf(d10, d11);
    float e00 = __expf(d00 - m0), e01 = __expf(d01 - m0);
    float e10 = __expf(d10 - m1), e11 = __expf(d11 - m1);
    float i0 = 1.f / (e00 + e01), i1 = 1.f / (e10 + e11);
    float p00 = 0.5f + 0.5f * e00 * i0, p01 = 0.5f + 0.5f * e01 * i0;  // blended softmax
    float p10 = 0.5f + 0.5f * e10 * i1, p11 = 0.5f + 0.5f * e11 * i1;
    for (int e = 0; e < 64; ++e) {
      float v0 = bf2f(B3[t0 * DMODEL + cb + e]), v1 = bf2f(B3[t1 * DMODEL + cb + e]);
      B0[t0 * DMODEL + cb + e] = f2bf(p00 * v0 + p01 * v1);
      B0[t1 * DMODEL + cb + e] = f2bf(p10 * v0 + p11 * v1);
    }
  }
  __syncthreads();

  // ---- attn_out = z @ W_O + b_O ----
  #pragma unroll 1
  for (int nt = 0; nt < 16; ++nt) {
    v8f ao = mm16_k8(B0, DMODEL, po, nt, lane);
    store_tile_lds(B1, DMODEL, nt, lane, ao, bO[nt * 16 + n], false);
  }
  __syncthreads();

  // ---- MLP fused: fc1+relu in 16x32 chunks feeding fc2 K-accumulation ----
  v8f acc2[16];
  #pragma unroll
  for (int i = 0; i < 16; ++i) acc2[i] = (v8f){0.f,0.f,0.f,0.f,0.f,0.f,0.f,0.f};
  #pragma unroll 1
  for (int c = 0; c < 32; ++c) {
    #pragma unroll
    for (int t2 = 0; t2 < 2; ++t2) {
      int nt1 = c * 2 + t2;
      v8f h = mm16_k8(B1, DMODEL, pf1, nt1, lane);
      store_tile_lds(HC, 32, t2, lane, h, fc1_b[nt1 * 16 + n], true);
    }
    __syncthreads();
    v16bf a = load_frag_a_lds(HC, 32, 0, lane);
    #pragma unroll
    for (int nt = 0; nt < 16; ++nt) {
      v16bf b = load_frag_b(pf2, nt * 32 + c, lane);   // fc2 has KC=32
      acc2[nt] = wmma_bf16(a, b, acc2[nt]);
    }
    __syncthreads();
  }
  // resid (bf16) to global workspace
  int r0 = (lane >> 4) * 8;
  #pragma unroll 1
  for (int nt = 0; nt < 16; ++nt) {
    float bias = fc2_b[nt * 16 + n];
    #pragma unroll
    for (int i = 0; i < 8; ++i) {
      int row = rowBase + r0 + i;
      resid[(size_t)row * DMODEL + nt * 16 + n] = f2bf(acc2[nt][i] + bias);
    }
  }
}

// ---------------- unembed: resid[N,256]bf16 @ un_w^T(256x1024 padded) + un_b ----------------
// Per-wave 16x256 bf16 resid tile = 8 KB *contiguous* in memory -> moved by one
// Tensor-Data-Mover descriptor (1-D tile: data_size=2B, tile_dim0=4096), completion
// fenced with s_wait_tensorcnt. A-fragments are then built with ds_load_b128.
__global__ __launch_bounds__(BLK) void unembed(
    const uint16_t* __restrict__ resid, const uint16_t* __restrict__ pun,
    const float* __restrict__ un_b, float* __restrict__ out) {
  __shared__ uint16_t stage[WAVES * 4096];   // 16 rows x 256 bf16 per wave
  int lane = threadIdx.x & 31, wave = threadIdx.x >> 5;
  int rowBase = blockIdx.x * ROWS_PER_BLK + wave * ROWS_PER_WAVE;

  // ---- TDM: DMA this wave's 8KB resid tile into LDS ----
  {
    uint64_t ga = (uint64_t)(uintptr_t)resid + (uint64_t)rowBase * (DMODEL * 2);
    uint32_t lds_off = (uint32_t)(wave * 4096 * 2);   // byte offset of wave's stage slice
    u32x4 g0;
    g0[0] = 0x1u;                                  // count=1, user descriptor
    g0[1] = lds_off;                               // lds_addr (bytes)
    g0[2] = (uint32_t)ga;                          // global_addr[31:0]
    g0[3] = ((uint32_t)(ga >> 32) & 0x01FFFFFFu)   // global_addr[56:32]
            | 0x80000000u;                         // type=2 ("image") at [127:126]
    i32x8 g1;
    g1[0] = (1 << 16);                             // data_size=1 -> 2-byte elements
    g1[1] = (int)(4096u << 16);                    // tensor_dim0[15:0] at bits [63:48]
    g1[2] = 0;                                     // tensor_dim0[31:16]=0, tensor_dim1 lo=0
    g1[3] = (int)(4096u << 16);                    // tile_dim0=4096 at bits [127:112]
    g1[4] = 0; g1[5] = 0; g1[6] = 0; g1[7] = 0;    // tile_dim1/2=0 (1-D), strides 0
    i32x4 gz4 = {0, 0, 0, 0};                      // groups 2/3 unused (<=2-D tensor)
    i32x8 gz8 = {0, 0, 0, 0, 0, 0, 0, 0};
    __builtin_amdgcn_tensor_load_to_lds(g0, g1, gz4, gz4, gz8, 0);
    __builtin_amdgcn_s_wait_tensorcnt(0);
    __asm__ volatile("" ::: "memory");             // keep ds loads below the wait
  }

  const uint16_t* aBase = stage + wave * 4096;
  v16bf afr[8];                       // all K fragments of the 16-row tile stay in VGPRs
  #pragma unroll
  for (int kc = 0; kc < 8; ++kc) afr[kc] = load_frag_a_lds(aBase, DMODEL, kc, lane);

  int col0 = lane & 15, r0 = (lane >> 4) * 8;
  #pragma unroll 1
  for (int nt = 0; nt < 64; ++nt) {
    __builtin_prefetch(pun + (size_t)(nt + 1) * 8 * 512, 0, 1);  // next tile's weights
    v8f acc = {0.f,0.f,0.f,0.f,0.f,0.f,0.f,0.f};
    #pragma unroll
    for (int kc = 0; kc < 8; ++kc) {
      v16bf b = load_frag_b(pun, nt * 8 + kc, lane);
      acc = wmma_bf16(afr[kc], b, acc);
    }
    int col = nt * 16 + col0;
    if (col < PVOC) {
      float bias = un_b[col];
      #pragma unroll
      for (int i = 0; i < 8; ++i) {
        size_t row = (size_t)(rowBase + r0 + i);
        out[row * PVOC + col] = acc[i] + bias;
      }
    }
  }
}

// ---------------- host launcher ----------------
extern "C" void kernel_launch(void* const* d_in, const int* in_sizes, int n_in,
                              void* d_out, int out_size, void* d_ws, size_t ws_size,
                              hipStream_t stream) {
  (void)in_sizes; (void)n_in; (void)out_size; (void)ws_size;
  const int*   x       = (const int*)  d_in[0];
  const float* tok_emb = (const float*)d_in[1];
  const float* pos_emb = (const float*)d_in[2];
  const float* W_Q = (const float*)d_in[3];
  const float* W_K = (const float*)d_in[4];
  const float* W_V = (const float*)d_in[5];
  const float* W_O = (const float*)d_in[6];
  const float* b_Q = (const float*)d_in[7];
  const float* b_K = (const float*)d_in[8];
  const float* b_V = (const float*)d_in[9];
  const float* b_O = (const float*)d_in[10];
  const float* fc1_w = (const float*)d_in[11];
  const float* fc1_b = (const float*)d_in[12];
  const float* fc2_w = (const float*)d_in[13];
  const float* fc2_b = (const float*)d_in[14];
  const float* un_w  = (const float*)d_in[15];
  const float* un_b  = (const float*)d_in[16];
  float* out = (float*)d_out;

  // workspace layout (bf16 elements): 4x 256x256 packs, 3x 256x1024-class packs, resid
  uint16_t* ws  = (uint16_t*)d_ws;
  uint16_t* pq  = ws;
  uint16_t* pkw = pq  + 65536;
  uint16_t* pvw = pkw + 65536;
  uint16_t* po  = pvw + 65536;
  uint16_t* pf1 = po  + 65536;          // 256 x 1024
  uint16_t* pf2 = pf1 + 262144;         // 1024 x 256
  uint16_t* pun = pf2 + 262144;         // 256 x 1024 (1009 zero-padded)
  uint16_t* resid = pun + 262144;       // 131072 x 256  (~64 MB)

  pack_w<<<256,  256, 0, stream>>>(W_Q,   pq,  256,  256, 0, 65536);
  pack_w<<<256,  256, 0, stream>>>(W_K,   pkw, 256,  256, 0, 65536);
  pack_w<<<256,  256, 0, stream>>>(W_V,   pvw, 256,  256, 0, 65536);
  pack_w<<<256,  256, 0, stream>>>(W_O,   po,  256,  256, 1, 65536);
  pack_w<<<1024, 256, 0, stream>>>(fc1_w, pf1, 256,  1024, 2, 262144);
  pack_w<<<1024, 256, 0, stream>>>(fc2_w, pf2, 1024, 256,  2, 262144);
  pack_w<<<1024, 256, 0, stream>>>(un_w,  pun, 256,  PVOC, 2, 262144);

  fused_fwd<<<NTOK / ROWS_PER_BLK, BLK, 0, stream>>>(
      x, tok_emb, pos_emb, b_Q, b_K, b_V, b_O, fc1_b, fc2_b,
      pq, pkw, pvw, po, pf1, pf2, resid);
  unembed<<<NTOK / ROWS_PER_BLK, BLK, 0, stream>>>(resid, pun, un_b, out);
}
